// SingleLayerFFNWithZones_64759516889945
// MI455X (gfx1250) — matrix operand order repaired and checked by reading, not compiled, not run on hardware
//
#include <hip/hip_runtime.h>

// MI455X / gfx1250 fused FFN + zone mask + softmax-confidence.
//   y = X @ W^T + b           (bf16 WMMA, f32 accumulate)
//   zoned = in-zone ? y : -inf (524 MB store -> the roofline limiter)
//   conf  = max(softmax(clamp(zoned)))  fused via fixed-ref exp + atomics.
// Pipeline: cvt f32->bf16 (W,X) once into ws, then the GEMM stages bf16
// tiles into LDS with global_load_async_to_lds_b128 (ASYNCcnt path).

typedef __attribute__((ext_vector_type(16))) __bf16 v16bf;
typedef __attribute__((ext_vector_type(8)))  float  v8f;
typedef int v4i_vec __attribute__((vector_size(16)));   // async builtin pointee

#define D_IN   1024
#define D_OUT  32000
#define NBATCH 4096
#define BM     128
#define BN     128
#define BK     32
#define LDSP   40            // padded LDS row stride in bf16 elems (80B, 16B-aligned)
#define KT     (D_IN / BK)   // 32
#define CLAMPV 25.0f

// Device pass only: host pass (x86) lacks amdgcn builtins but still parses
// kernel bodies, so it must see the fallback body.
#if defined(__HIP_DEVICE_COMPILE__) &&                                  \
    __has_builtin(__builtin_amdgcn_global_load_async_to_lds_b128) &&    \
    __has_builtin(__builtin_amdgcn_s_wait_asynccnt)
#define HAVE_ASYNC 1
#else
#define HAVE_ASYNC 0
#endif

#define GPTR(p) ((__attribute__((address_space(1))) v4i_vec*)(p))
#define LPTR(p) ((__attribute__((address_space(3))) v4i_vec*)(p))

__device__ __forceinline__ unsigned int pack2bf16(float a, float b) {
  union { __bf16 h[2]; unsigned int u; } r;
  r.h[0] = (__bf16)a;   // RNE; lowers to v_cvt_pk_bf16_f32
  r.h[1] = (__bf16)b;
  return r.u;
}

__global__ void __launch_bounds__(256)
cvt_bf16(const float4* __restrict__ src, uint2* __restrict__ dst, int n4) {
  int i = blockIdx.x * blockDim.x + threadIdx.x;
  if (i < n4) {
    float4 v = src[i];
    uint2 p;
    p.x = pack2bf16(v.x, v.y);
    p.y = pack2bf16(v.z, v.w);
    dst[i] = p;
  }
}

// ---- Shared epilogue: bias + zone mask + zoned store + confidence partials.
__device__ __forceinline__ void
zone_epilogue(v8f acc[4][2], const int* zsL, const int* zeL,
              float* ssum, unsigned int* smax,
              const float* __restrict__ bvec, float* __restrict__ zoned,
              float* __restrict__ rowsum, unsigned int* __restrict__ rowemax,
              int m0, int n0, int wm, int wn, int hf, int ln, int t) {
  const int   col0 = n0 + wn * 32 + ln;
  const int   col1 = col0 + 16;
  const float bv0  = bvec[col0];
  const float bv1  = bvec[col1];
  const float NINF = -__builtin_inff();

#pragma unroll
  for (int mi = 0; mi < 4; ++mi) {
#pragma unroll
    for (int r = 0; r < 8; ++r) {
      const int lrow = wm * 64 + mi * 16 + hf * 8 + r;  // C layout: VGPR r -> M=r / 8+r
      const int zs = zsL[lrow];
      const int ze = zeL[lrow];
      const float y0 = acc[mi][0][r] + bv0;
      const float y1 = acc[mi][1][r] + bv1;
      const bool in0 = (col0 >= zs) && (col0 < ze);
      const bool in1 = (col1 >= zs) && (col1 < ze);
      const size_t ob = (size_t)(m0 + lrow) * D_OUT;
      zoned[ob + col0] = in0 ? y0 : NINF;
      zoned[ob + col1] = in1 ? y1 : NINF;
      const float c0 = in0 ? fminf(fmaxf(y0, -CLAMPV), CLAMPV) : -CLAMPV;
      const float c1 = in1 ? fminf(fmaxf(y1, -CLAMPV), CLAMPV) : -CLAMPV;
      float mx = fmaxf(c0, c1);
      float es = __expf(c0 - CLAMPV) + __expf(c1 - CLAMPV);
#pragma unroll
      for (int s = 1; s < 16; s <<= 1) {   // reduce across 16-lane half (one row)
        mx  = fmaxf(mx, __shfl_xor(mx, s));
        es += __shfl_xor(es, s);
      }
      if (ln == 0) {
        atomicMax(&smax[lrow], __float_as_uint(__expf(mx - CLAMPV)));
        atomicAdd(&ssum[lrow], es);
      }
    }
  }
  __syncthreads();
  if (t < BM) {
    atomicAdd(&rowsum[m0 + t], ssum[t]);
    atomicMax(&rowemax[m0 + t], smax[t]);   // exp>=0: uint bits order-preserving
  }
}

// ---- Fast path: bf16 operands in workspace, async-to-LDS staging.
// (256,1): allow a full register budget so acc/fragment/address registers get
// disjoint homes (removes v_mov_b64 acc copies + hazard NOPs in the K-loop).
__global__ void __launch_bounds__(256, 1)
ffn_zone_gemm_bf16(const int* __restrict__ zones, const float* __restrict__ bvec,
                   const unsigned short* __restrict__ Xb,
                   const unsigned short* __restrict__ Wb,
                   float* __restrict__ zoned, float* __restrict__ rowsum,
                   unsigned int* __restrict__ rowemax) {
  __shared__ __align__(16) unsigned short Ab[2][BM * LDSP];
  __shared__ __align__(16) unsigned short Bb[2][BN * LDSP];
  __shared__ int   zsL[BM];
  __shared__ int   zeL[BM];
  __shared__ float ssum[BM];
  __shared__ unsigned int smax[BM];

  const int t    = threadIdx.x;
  const int lane = t & 31;
  const int wv   = t >> 5;       // 8 waves: 2 (M) x 4 (N)
  const int wm   = wv >> 2;
  const int wn   = wv & 3;
  const int hf   = lane >> 4;    // lane half -> K-half select (this code's contiguous k = 16*hf + s order)
  const int ln   = lane & 15;
  const int n0   = blockIdx.x * BN;
  const int m0   = blockIdx.y * BM;

  if (t < BM) {
    zsL[t]  = zones[(m0 + t) * 2 + 0];
    zeL[t]  = zones[(m0 + t) * 2 + 1];
    ssum[t] = 0.0f;
    smax[t] = 0u;
  }

  const v8f vzero = {0.f, 0.f, 0.f, 0.f, 0.f, 0.f, 0.f, 0.f};
  v8f acc[4][2];
#pragma unroll
  for (int mi = 0; mi < 4; ++mi) { acc[mi][0] = vzero; acc[mi][1] = vzero; }

  union U { v16bf v; uint4 q[2]; };
  auto do_compute = [&](int cur) {
    U b0, b1;
    {
      const uint4* pb0 = (const uint4*)&Bb[cur][(wn * 32 + ln) * LDSP];
      b0.q[0] = pb0[2 * hf]; b0.q[1] = pb0[2 * hf + 1];
      const uint4* pb1 = (const uint4*)&Bb[cur][(wn * 32 + 16 + ln) * LDSP];
      b1.q[0] = pb1[2 * hf]; b1.q[1] = pb1[2 * hf + 1];
    }
#pragma unroll
    for (int mi = 0; mi < 4; ++mi) {
      U a;
      const uint4* pa = (const uint4*)&Ab[cur][(wm * 64 + mi * 16 + ln) * LDSP];
      a.q[0] = pa[2 * hf]; a.q[1] = pa[2 * hf + 1];
      acc[mi][0] = __builtin_amdgcn_wmma_f32_16x16x32_bf16(
          false, a.v, false, b0.v, (short)0, acc[mi][0], false, false);
      acc[mi][1] = __builtin_amdgcn_wmma_f32_16x16x32_bf16(
          false, a.v, false, b1.v, (short)0, acc[mi][1], false, false);
    }
  };

#if HAVE_ASYNC
  auto stage = [&](int kt, int buf) {
    const int k0 = kt * BK;
#pragma unroll
    for (int i = 0; i < 2; ++i) {
      int id  = t + i * 256;
      int row = id >> 2;    // 4 x 16B chunks per 32-elem row
      int cq  = id & 3;
      __builtin_amdgcn_global_load_async_to_lds_b128(
          GPTR(Xb + (size_t)(m0 + row) * D_IN + k0 + cq * 8),
          LPTR(&Ab[buf][row * LDSP + cq * 8]), 0, 0);
      __builtin_amdgcn_global_load_async_to_lds_b128(
          GPTR(Wb + (size_t)(n0 + row) * D_IN + k0 + cq * 8),
          LPTR(&Bb[buf][row * LDSP + cq * 8]), 0, 0);
    }
  };
  auto wait_stage = [&]() { __builtin_amdgcn_s_wait_asynccnt(0); };
#else
  // Host-pass parse body / toolchains without async builtins.
  auto stage = [&](int kt, int buf) {
    const int k0 = kt * BK;
#pragma unroll
    for (int i = 0; i < 2; ++i) {
      int id  = t + i * 256;
      int row = id >> 2;
      int cq  = id & 3;
      *(uint4*)&Ab[buf][row * LDSP + cq * 8] =
          *(const uint4*)(Xb + (size_t)(m0 + row) * D_IN + k0 + cq * 8);
      *(uint4*)&Bb[buf][row * LDSP + cq * 8] =
          *(const uint4*)(Wb + (size_t)(n0 + row) * D_IN + k0 + cq * 8);
    }
  };
  auto wait_stage = [&]() {};
#endif

  stage(0, 0);
  wait_stage();
  __syncthreads();
  for (int kt = 0; kt < KT; ++kt) {
    const int cur = kt & 1;
    if (kt + 1 < KT) stage(kt + 1, cur ^ 1);   // overlap copies with compute
    do_compute(cur);
    if (kt + 1 < KT) {
      wait_stage();
      __syncthreads();
    }
  }

  zone_epilogue(acc, zsL, zeL, ssum, smax, bvec, zoned, rowsum, rowemax,
                m0, n0, wm, wn, hf, ln, t);
}

// ---- Fallback (workspace too small): f32 loads, v_cvt_pk_bf16_f32 in-loop.
__global__ void __launch_bounds__(256, 1)
ffn_zone_gemm_f32(const float* __restrict__ X, const int* __restrict__ zones,
                  const float* __restrict__ W, const float* __restrict__ bvec,
                  float* __restrict__ zoned, float* __restrict__ rowsum,
                  unsigned int* __restrict__ rowemax) {
  __shared__ __align__(16) unsigned short Ab[2][BM * LDSP];
  __shared__ __align__(16) unsigned short Bb[2][BN * LDSP];
  __shared__ int   zsL[BM];
  __shared__ int   zeL[BM];
  __shared__ float ssum[BM];
  __shared__ unsigned int smax[BM];

  const int t    = threadIdx.x;
  const int lane = t & 31;
  const int wv   = t >> 5;
  const int wm   = wv >> 2;
  const int wn   = wv & 3;
  const int hf   = lane >> 4;
  const int ln   = lane & 15;
  const int n0   = blockIdx.x * BN;
  const int m0   = blockIdx.y * BM;

  if (t < BM) {
    zsL[t]  = zones[(m0 + t) * 2 + 0];
    zeL[t]  = zones[(m0 + t) * 2 + 1];
    ssum[t] = 0.0f;
    smax[t] = 0u;
  }

  const v8f vzero = {0.f, 0.f, 0.f, 0.f, 0.f, 0.f, 0.f, 0.f};
  v8f acc[4][2];
#pragma unroll
  for (int mi = 0; mi < 4; ++mi) { acc[mi][0] = vzero; acc[mi][1] = vzero; }

  union U { v16bf v; uint4 q[2]; };
  auto do_compute = [&](int cur) {
    U b0, b1;
    {
      const uint4* pb0 = (const uint4*)&Bb[cur][(wn * 32 + ln) * LDSP];
      b0.q[0] = pb0[2 * hf]; b0.q[1] = pb0[2 * hf + 1];
      const uint4* pb1 = (const uint4*)&Bb[cur][(wn * 32 + 16 + ln) * LDSP];
      b1.q[0] = pb1[2 * hf]; b1.q[1] = pb1[2 * hf + 1];
    }
#pragma unroll
    for (int mi = 0; mi < 4; ++mi) {
      U a;
      const uint4* pa = (const uint4*)&Ab[cur][(wm * 64 + mi * 16 + ln) * LDSP];
      a.q[0] = pa[2 * hf]; a.q[1] = pa[2 * hf + 1];
      acc[mi][0] = __builtin_amdgcn_wmma_f32_16x16x32_bf16(
          false, a.v, false, b0.v, (short)0, acc[mi][0], false, false);
      acc[mi][1] = __builtin_amdgcn_wmma_f32_16x16x32_bf16(
          false, a.v, false, b1.v, (short)0, acc[mi][1], false, false);
    }
  };

  float4 ra[4], rb[4];
  auto g_load = [&](int kt) {
    const int k0 = kt * BK;
#pragma unroll
    for (int i = 0; i < 4; ++i) {
      int id  = t + i * 256;
      int row = id >> 3;
      int cq  = id & 7;
      ra[i] = *(const float4*)(X + (size_t)(m0 + row) * D_IN + k0 + cq * 4);
      rb[i] = *(const float4*)(W + (size_t)(n0 + row) * D_IN + k0 + cq * 4);
    }
  };
  auto l_store = [&](int buf) {
#pragma unroll
    for (int i = 0; i < 4; ++i) {
      int id  = t + i * 256;
      int row = id >> 3;
      int cq  = id & 7;
      uint2 pa; pa.x = pack2bf16(ra[i].x, ra[i].y); pa.y = pack2bf16(ra[i].z, ra[i].w);
      uint2 pb; pb.x = pack2bf16(rb[i].x, rb[i].y); pb.y = pack2bf16(rb[i].z, rb[i].w);
      *(uint2*)&Ab[buf][row * LDSP + cq * 4] = pa;
      *(uint2*)&Bb[buf][row * LDSP + cq * 4] = pb;
    }
  };

  g_load(0); l_store(0); __syncthreads();
  for (int kt = 0; kt < KT; ++kt) {
    const int cur = kt & 1;
    if (kt + 1 < KT) g_load(kt + 1);
    do_compute(cur);
    if (kt + 1 < KT) { l_store(cur ^ 1); __syncthreads(); }
  }

  zone_epilogue(acc, zsL, zeL, ssum, smax, bvec, zoned, rowsum, rowemax,
                m0, n0, wm, wn, hf, ln, t);
}

__global__ void init_stats(float* rowsum, unsigned int* rowemax) {
  int i = blockIdx.x * blockDim.x + threadIdx.x;
  if (i < NBATCH) { rowsum[i] = 0.0f; rowemax[i] = 0u; }
}

__global__ void finalize_conf(const float* __restrict__ rowsum,
                              const unsigned int* __restrict__ rowemax,
                              float* __restrict__ conf) {
  int i = blockIdx.x * blockDim.x + threadIdx.x;
  if (i < NBATCH) {
    conf[i] = __uint_as_float(rowemax[i]) / rowsum[i];  // max exp / sum exp
  }
}

extern "C" void kernel_launch(void* const* d_in, const int* in_sizes, int n_in,
                              void* d_out, int out_size, void* d_ws, size_t ws_size,
                              hipStream_t stream) {
  const float* X     = (const float*)d_in[0];   // [4096,1024]  f32
  const int*   zones = (const int*)d_in[1];     // [4096,2]     i32
  const float* W     = (const float*)d_in[2];   // [32000,1024] f32
  const float* bv    = (const float*)d_in[3];   // [32000]      f32

  float* zoned = (float*)d_out;                           // [4096,32000]
  float* conf  = (float*)d_out + (size_t)NBATCH * D_OUT;  // [4096]

  const size_t WB_BYTES    = (size_t)D_OUT * D_IN * 2;    // 65.5 MB
  const size_t XB_BYTES    = (size_t)NBATCH * D_IN * 2;   //  8.4 MB
  const size_t STATS_BYTES = (size_t)NBATCH * 8;          // 32 KB
  const int big = (ws_size >= WB_BYTES + XB_BYTES + STATS_BYTES) ? 1 : 0;

  unsigned short* Wb = (unsigned short*)d_ws;
  unsigned short* Xb = (unsigned short*)((char*)d_ws + WB_BYTES);
  float* rowsum = big ? (float*)((char*)d_ws + WB_BYTES + XB_BYTES)
                      : (float*)d_ws;
  unsigned int* rowemax = (unsigned int*)(rowsum + NBATCH);

  init_stats<<<(NBATCH + 255) / 256, 256, 0, stream>>>(rowsum, rowemax);

  dim3 grid(D_OUT / BN, NBATCH / BM);   // 250 x 32
  if (big) {
    const int wN4 = (D_OUT * D_IN) / 4;   // 8.192M float4
    const int xN4 = (NBATCH * D_IN) / 4;  // 1.049M float4
    cvt_bf16<<<(wN4 + 255) / 256, 256, 0, stream>>>((const float4*)W, (uint2*)Wb, wN4);
    cvt_bf16<<<(xN4 + 255) / 256, 256, 0, stream>>>((const float4*)X, (uint2*)Xb, xN4);
    ffn_zone_gemm_bf16<<<grid, 256, 0, stream>>>(zones, bv, Xb, Wb,
                                                 zoned, rowsum, rowemax);
  } else {
    ffn_zone_gemm_f32<<<grid, 256, 0, stream>>>(X, zones, W, bv,
                                                zoned, rowsum, rowemax);
  }

  finalize_conf<<<(NBATCH + 255) / 256, 256, 0, stream>>>(rowsum, rowemax, conf);
}
